// NSTLoss_85555748537091
// MI455X (gfx1250) — matrix operand order repaired and checked by reading, not compiled, not run on hardware
//
#include <hip/hip_runtime.h>
#include <hip/hip_bf16.h>

typedef __attribute__((ext_vector_type(16))) __bf16 v16bf;
typedef __attribute__((ext_vector_type(8)))  __bf16 v8bf;
typedef __attribute__((ext_vector_type(8)))  float  v8f;
typedef __attribute__((ext_vector_type(4)))  float  v4f;

#define BATCH  16
#define CCH    128
#define TLEN   16384
#define KSLICE 512
#define KC     32
#define NCH    (KSLICE / KC)
#define LDK    40            // padded bf16 row stride (conflict-free, 16B aligned)
#define PLANE  (CCH * LDK)   // 5120 bf16 per plane
#define NGRAM  (3 * BATCH * CCH * CCH)

// ---- operand builders: contiguous k order (k = 16*(lane/16) + s) in A and B (wave32) -------------
// A operand 16x32 bf16: lane<16 holds row m=lane, K 0..15;
// lane>=16 holds row m=lane-16, K 16..31.
__device__ __forceinline__ v16bf build_a(const __bf16* plane, int rowBase, int lane) {
    int m  = rowBase + (lane & 15);
    int kh = (lane & 16) ? 16 : 0;
    const __bf16* q = plane + m * LDK + kh;
    v8bf r0 = *(const v8bf*)(q);          // K = kh .. kh+7
    v8bf r1 = *(const v8bf*)(q + 8);      // K = kh+8 .. kh+15
    return __builtin_shufflevector(r0, r1, 0,1,2,3,4,5,6,7,8,9,10,11,12,13,14,15);
}

// B operand 32x16 bf16: lane<16 holds col n=lane, K 0..15; lane>=16 K 16..31.
__device__ __forceinline__ v16bf build_b(const __bf16* plane, int colBase, int lane) {
    int n  = colBase + (lane & 15);
    int kb = (lane & 16) ? 16 : 0;
    const __bf16* q = plane + n * LDK + kb;
    v8bf r0 = *(const v8bf*)(q);          // K = kb .. kb+7
    v8bf r1 = *(const v8bf*)(q + 8);      // K = kb+8 .. kb+15
    return __builtin_shufflevector(r0, r1, 0,1,2,3,4,5,6,7,8,9,10,11,12,13,14,15);
}

// bf16x3 split accumulation: c += hi*hi + hi*lo + lo*hi  (lo*lo ~ 2^-32, dropped)
__device__ __forceinline__ v8f wmma3(v16bf ah, v16bf al, v16bf bh, v16bf bl, v8f c) {
    c = __builtin_amdgcn_wmma_f32_16x16x32_bf16(false, ah, false, bh, (short)0, c, false, false);
    c = __builtin_amdgcn_wmma_f32_16x16x32_bf16(false, ah, false, bl, (short)0, c, false, false);
    c = __builtin_amdgcn_wmma_f32_16x16x32_bf16(false, al, false, bh, (short)0, c, false, false);
    return c;
}

__device__ __forceinline__ void cvt8(v4f a, v4f b, v8bf& h, v8bf& l) {
#pragma unroll
    for (int i = 0; i < 4; ++i) {
        float x = a[i]; __bf16 hx = (__bf16)x; h[i]     = hx; l[i]     = (__bf16)(x - (float)hx);
        float y = b[i]; __bf16 hy = (__bf16)y; h[i + 4] = hy; l[i + 4] = (__bf16)(y - (float)hy);
    }
}

// ---- kernel 0: zero gram workspace + output -------------------------------
__global__ void zero_kernel(float* ws, float* out, int n) {
    int i = blockIdx.x * blockDim.x + threadIdx.x;
    if (i < n) ws[i] = 0.0f;
    if (i == 0) out[0] = 0.0f;
}

// ---- kernel 1: raw batched Grams via bf16x3 WMMA --------------------------
// grid = (TLEN/KSLICE, BATCH), block = 512 (16 waves, 4x4 wave grid)
__global__ __launch_bounds__(512) void gram_kernel(const float* __restrict__ fs,
                                                   const float* __restrict__ ft,
                                                   float* __restrict__ gram) {
    __shared__ __bf16 lds[4 * PLANE];   // planes: hiS, loS, hiT, loT (40 KiB)

    const int tid  = threadIdx.x;
    const int lane = tid & 31;
    const int wave = tid >> 5;
    const int r0   = (wave >> 2) * 32;  // wave's output row base
    const int c0   = (wave & 3) * 32;   // wave's output col base
    const int b    = blockIdx.y;
    const size_t kbase = (size_t)blockIdx.x * KSLICE;

    const float* fsb = fs + (size_t)b * CCH * TLEN;
    const float* ftb = ft + (size_t)b * CCH * TLEN;

    v8f zero = {0.f,0.f,0.f,0.f,0.f,0.f,0.f,0.f};
    v8f accss[2][2], acctt[2][2], accst[2][2];
#pragma unroll
    for (int i = 0; i < 2; ++i)
#pragma unroll
        for (int j = 0; j < 2; ++j) { accss[i][j] = zero; acctt[i][j] = zero; accst[i][j] = zero; }

    // loader assignment: 256 threads per matrix, 2 threads per row, 16 floats each
    const int ms   = tid >> 8;          // 0 -> f_s, 1 -> f_t
    const int t2   = tid & 255;
    const int lrow = t2 >> 1;
    const int lcol = (t2 & 1) * 16;
    const float* src = (ms ? ftb : fsb) + (size_t)lrow * TLEN + lcol;
    __bf16* hiP = lds + (ms * 2) * PLANE + lrow * LDK + lcol;
    __bf16* loP = hiP + PLANE;

    const __bf16* hiS = lds;
    const __bf16* loS = lds + PLANE;
    const __bf16* hiT = lds + 2 * PLANE;
    const __bf16* loT = lds + 3 * PLANE;

    // software pipeline: prefetch chunk 0 into registers
    v4f x0, x1, x2, x3;
    {
        const v4f* g = (const v4f*)(src + kbase);
        x0 = g[0]; x1 = g[1]; x2 = g[2]; x3 = g[3];
    }

#pragma unroll 1
    for (int ch = 0; ch < NCH; ++ch) {
        __syncthreads();                 // previous compute done before LDS overwrite
        // ---- convert prefetched fp32 -> bf16 hi/lo, store once per element
        v8bf h0, h1, l0, l1;
        cvt8(x0, x1, h0, l0);
        cvt8(x2, x3, h1, l1);
        *(v8bf*)(hiP)     = h0;  *(v8bf*)(hiP + 8) = h1;
        *(v8bf*)(loP)     = l0;  *(v8bf*)(loP + 8) = l1;
        __syncthreads();

        // ---- prefetch next chunk: HBM latency hidden behind this chunk's WMMAs
        if (ch + 1 < NCH) {              // wave-uniform branch, EXEC stays all-ones
            const v4f* g = (const v4f*)(src + kbase + (size_t)(ch + 1) * KC);
            x0 = g[0]; x1 = g[1]; x2 = g[2]; x3 = g[3];
        }

        // ---- S x S^T
        v16bf ah0 = build_a(hiS, r0,      lane), ah1 = build_a(hiS, r0 + 16, lane);
        v16bf al0 = build_a(loS, r0,      lane), al1 = build_a(loS, r0 + 16, lane);
        v16bf bh0 = build_b(hiS, c0,      lane), bh1 = build_b(hiS, c0 + 16, lane);
        v16bf bl0 = build_b(loS, c0,      lane), bl1 = build_b(loS, c0 + 16, lane);
        accss[0][0] = wmma3(ah0, al0, bh0, bl0, accss[0][0]);
        accss[0][1] = wmma3(ah0, al0, bh1, bl1, accss[0][1]);
        accss[1][0] = wmma3(ah1, al1, bh0, bl0, accss[1][0]);
        accss[1][1] = wmma3(ah1, al1, bh1, bl1, accss[1][1]);

        // ---- S x T^T (reuse A from S, rebuild B from T)
        bh0 = build_b(hiT, c0, lane);  bh1 = build_b(hiT, c0 + 16, lane);
        bl0 = build_b(loT, c0, lane);  bl1 = build_b(loT, c0 + 16, lane);
        accst[0][0] = wmma3(ah0, al0, bh0, bl0, accst[0][0]);
        accst[0][1] = wmma3(ah0, al0, bh1, bl1, accst[0][1]);
        accst[1][0] = wmma3(ah1, al1, bh0, bl0, accst[1][0]);
        accst[1][1] = wmma3(ah1, al1, bh1, bl1, accst[1][1]);

        // ---- T x T^T (reuse B from T, rebuild A from T)
        ah0 = build_a(hiT, r0, lane);  ah1 = build_a(hiT, r0 + 16, lane);
        al0 = build_a(loT, r0, lane);  al1 = build_a(loT, r0 + 16, lane);
        acctt[0][0] = wmma3(ah0, al0, bh0, bl0, acctt[0][0]);
        acctt[0][1] = wmma3(ah0, al0, bh1, bl1, acctt[0][1]);
        acctt[1][0] = wmma3(ah1, al1, bh0, bl0, acctt[1][0]);
        acctt[1][1] = wmma3(ah1, al1, bh1, bl1, acctt[1][1]);
    }

    // ---- accumulate partial Grams: C/D layout lane n = lane&15, m = v + 8*(lane>>4)
    const int mo = (lane >> 4) * 8;
    const int n  = lane & 15;
#pragma unroll
    for (int g = 0; g < 3; ++g) {
        v8f (*acc)[2] = (g == 0) ? accss : ((g == 1) ? acctt : accst);
        float* base = gram + ((size_t)(g * BATCH + b)) * CCH * CCH;
#pragma unroll
        for (int i = 0; i < 2; ++i)
#pragma unroll
            for (int j = 0; j < 2; ++j) {
                float* t = base + (size_t)(r0 + 16 * i + mo) * CCH + (c0 + 16 * j + n);
#pragma unroll
                for (int v = 0; v < 8; ++v)
                    atomicAdd(t + (size_t)v * CCH, acc[i][j][v]);
            }
    }
}

// ---- kernel 2: normalize by Gram diagonals, reduce to scalar loss ---------
__global__ __launch_bounds__(256) void loss_kernel(const float* __restrict__ gram,
                                                   float* __restrict__ out) {
    const int b = blockIdx.x;
    const float* Gss = gram + ((size_t)(0 * BATCH + b)) * CCH * CCH;
    const float* Gtt = gram + ((size_t)(1 * BATCH + b)) * CCH * CCH;
    const float* Gst = gram + ((size_t)(2 * BATCH + b)) * CCH * CCH;

    __shared__ float invS[CCH], invT[CCH];
    for (int c = threadIdx.x; c < CCH; c += blockDim.x) {
        float ns = sqrtf(fmaxf(Gss[c * CCH + c], 0.0f));
        float nt = sqrtf(fmaxf(Gtt[c * CCH + c], 0.0f));
        invS[c] = 1.0f / fmaxf(ns, 1e-12f);   // F.normalize eps semantics
        invT[c] = 1.0f / fmaxf(nt, 1e-12f);
    }
    __syncthreads();

    float partial = 0.0f;
    for (int idx = threadIdx.x; idx < CCH * CCH; idx += blockDim.x) {
        int c = idx >> 7, d = idx & 127;
        float gss = Gss[idx] * invS[c] * invS[d];
        float gtt = Gtt[idx] * invT[c] * invT[d];
        float gst = Gst[idx] * invS[c] * invT[d];
        partial += gtt * gtt + gss * gss - 2.0f * gst * gst;
    }

    __shared__ float red[256];
    red[threadIdx.x] = partial;
    __syncthreads();
    for (int s = 128; s > 0; s >>= 1) {
        if (threadIdx.x < s) red[threadIdx.x] += red[threadIdx.x + s];
        __syncthreads();
    }
    if (threadIdx.x == 0)
        atomicAdd(out, red[0] * (1.0f / (float)(BATCH * CCH * CCH)));
}

extern "C" void kernel_launch(void* const* d_in, const int* in_sizes, int n_in,
                              void* d_out, int out_size, void* d_ws, size_t ws_size,
                              hipStream_t stream) {
    (void)in_sizes; (void)n_in; (void)out_size; (void)ws_size;
    const float* fs = (const float*)d_in[0];
    const float* ft = (const float*)d_in[1];
    float* out  = (float*)d_out;
    float* gram = (float*)d_ws;   // 3 * 16 * 128 * 128 fp32 = 3 MiB

    zero_kernel<<<(NGRAM + 255) / 256, 256, 0, stream>>>(gram, out, NGRAM);

    dim3 grid(TLEN / KSLICE, BATCH);   // 32 K-slices x 16 batches = 512 WGs
    gram_kernel<<<grid, 512, 0, stream>>>(fs, ft, gram);

    loss_kernel<<<BATCH, 256, 0, stream>>>(gram, out);
}
